// FlashMHA_82265803588274
// MI455X (gfx1250) — matrix operand order repaired and checked by reading, not compiled, not run on hardware
//
#include <hip/hip_runtime.h>

// ---------------------------------------------------------------------------
// Fused RoPE multi-head attention for MI455X (gfx1250, wave32, WMMA).
// B=2, S=3072, C=256, heads=8, d=32.  All GEMMs via v_wmma_f32_16x16x32_f16.
// ---------------------------------------------------------------------------

#define C_DIM   256
#define NHEADS  8
#define HDIM    32
#define SEQ     3072
#define BATCH   2
#define MROWS   (BATCH * SEQ)      // 6144
#define QTILES  (SEQ / 16)         // 192

typedef __attribute__((ext_vector_type(16))) _Float16 v16h;
typedef __attribute__((ext_vector_type(8)))  _Float16 v8h;
typedef __attribute__((ext_vector_type(8)))  float    v8f;

union HF16 { v16h v; v8h h[2]; };

__device__ __forceinline__ v8f wmma_f16(v16h a, v16h b, v8f c) {
    // D = A(16x32 f16) * B(32x16 f16) + C(16x16 f32)
    return __builtin_amdgcn_wmma_f32_16x16x32_f16(
        /*neg_a=*/false, a, /*neg_b=*/false, b,
        /*c_mod=*/(short)0, c, /*reuse_a=*/false, /*reuse_b=*/false);
}

// A-operand fragment (16x32 f16): lane (m = l%16, kh = l/16) holds
// K in [8*kh, 8*kh+8) and [16+8*kh, 16+8*kh+8)  -> two 16B loads per lane.
__device__ __forceinline__ v16h load_a_frag(const _Float16* row, int kh) {
    HF16 u;
    u.h[0] = *(const v8h*)(row + 8 * kh);
    u.h[1] = *(const v8h*)(row + 16 + 8 * kh);
    return u.v;
}

// B-operand fragment (32x16 f16): lane (n = l%16, kh = l/16) holds
// K in [8*kh, 8*kh+8) and [16+8*kh, 16+8*kh+8)  -> two 16B loads per lane.
__device__ __forceinline__ v16h load_b_frag(const _Float16* row, int kh) {
    HF16 u;
    u.h[0] = *(const v8h*)(row + 8 * kh);
    u.h[1] = *(const v8h*)(row + 16 + 8 * kh);
    return u.v;
}

// ---------------------------------------------------------------------------
// Kernel 0: f32 -> f16 convert
// ---------------------------------------------------------------------------
__global__ void cvt_f16_kernel(const float* __restrict__ src,
                               _Float16* __restrict__ dst, int n) {
    int i = blockIdx.x * 256 + threadIdx.x;
    if (i < n) dst[i] = (_Float16)src[i];
}

// ---------------------------------------------------------------------------
// Kernel 1: QKV projection + bias + RoPE (+ 1/sqrt(d) pre-scale of Q).
// grid = (MROWS/16, 1, 3 [q,k,v]); block = 256 (8 waves, one head per wave).
// Each wave computes a 16(s) x 32(head) tile: RoPE pair (j, j+16) lives in
// the same lane/VGPR of acc0/acc1.
// ---------------------------------------------------------------------------
__global__ __launch_bounds__(256) void qkv_rope_kernel(
    const _Float16* __restrict__ xh,
    const _Float16* __restrict__ wq,
    const _Float16* __restrict__ wk,
    const _Float16* __restrict__ wv,
    const float* __restrict__ q_bias,
    const float* __restrict__ v_bias,
    _Float16* __restrict__ Qo,   // [B, h, S, d]
    _Float16* __restrict__ Ko,   // [B, h, S, d]
    _Float16* __restrict__ Vt)   // [B, h, d, S]
{
    const int z    = blockIdx.z;             // 0=q 1=k 2=v
    const int row0 = blockIdx.x * 16;        // tile of 16 sequence rows
    const int b    = row0 / SEQ;
    const int s0   = row0 % SEQ;
    const int head = threadIdx.x >> 5;       // wave id = head
    const int lane = threadIdx.x & 31;
    const int ln   = lane & 15;
    const int kh   = lane >> 4;
    const int n0   = head * HDIM;            // head's column base in C

    const _Float16* W = (z == 0) ? wq : (z == 1) ? wk : wv;

    const _Float16* arow  = xh + (size_t)(row0 + ln) * C_DIM;
    const _Float16* brow0 = W + (size_t)(n0 + ln) * C_DIM;        // cols j=0..15
    const _Float16* brow1 = W + (size_t)(n0 + 16 + ln) * C_DIM;   // cols j=16..31

    v8f acc0 = {}, acc1 = {};
#pragma unroll
    for (int k0 = 0; k0 < C_DIM; k0 += 32) {
        v16h A  = load_a_frag(arow + k0, kh);
        v16h B0 = load_b_frag(brow0 + k0, kh);
        v16h B1 = load_b_frag(brow1 + k0, kh);
        acc0 = wmma_f16(A, B0, acc0);
        acc1 = wmma_f16(A, B1, acc1);
    }

    // bias (q and v only)
    if (z != 1) {
        const float* bias = (z == 0) ? q_bias : v_bias;
        float b0 = bias[n0 + ln];
        float b1 = bias[n0 + 16 + ln];
#pragma unroll
        for (int r = 0; r < 8; ++r) { acc0[r] += b0; acc1[r] += b1; }
    }

    if (z < 2) {
        // RoPE: j = ln (0..15), pair partner j+16 in acc1; s varies per row.
        const float invf = __expf(-(float)ln * 0.57564627324851f); // ln(1e4)/16
        const float qscale = (z == 0) ? 0.17677669529663687f : 1.0f; // 1/sqrt(32)
#pragma unroll
        for (int r = 0; r < 8; ++r) {
            float s   = (float)(s0 + r + 8 * kh);
            float ang = s * invf;
            float cs  = __cosf(ang);
            float sn  = __sinf(ang);
            float lo = acc0[r], hi = acc1[r];
            acc0[r] = (lo * cs - hi * sn) * qscale;
            acc1[r] = (hi * cs + lo * sn) * qscale;
        }
        _Float16* O = (z == 0) ? Qo : Ko;
        size_t base = (size_t)(b * NHEADS + head) * SEQ;
#pragma unroll
        for (int r = 0; r < 8; ++r) {
            int srow = s0 + r + 8 * kh;
            _Float16* p = O + (base + srow) * HDIM;
            p[ln]      = (_Float16)acc0[r];
            p[16 + ln] = (_Float16)acc1[r];
        }
    } else {
        // V stored transposed: Vt[b,h,j,s]
        size_t vbase = (size_t)(b * NHEADS + head) * HDIM * SEQ;
#pragma unroll
        for (int r = 0; r < 8; ++r) {
            int srow = s0 + r + 8 * kh;
            Vt[vbase + (size_t)ln * SEQ + srow]        = (_Float16)acc0[r];
            Vt[vbase + (size_t)(16 + ln) * SEQ + srow] = (_Float16)acc1[r];
        }
    }
}

// ---------------------------------------------------------------------------
// Kernel 2: flash attention.  grid = B*h*(S/16) single-wave blocks.
// Per chunk of 32 keys: 2 score WMMAs, online softmax (shfl_xor reductions),
// P transposed D->A layout through LDS, 2 PV WMMAs.
// ---------------------------------------------------------------------------
__global__ __launch_bounds__(32) void flash_attn_kernel(
    const _Float16* __restrict__ Q,   // [B,h,S,d] (pre-scaled by 1/sqrt(d))
    const _Float16* __restrict__ K,   // [B,h,S,d]
    const _Float16* __restrict__ Vt,  // [B,h,d,S]
    _Float16* __restrict__ ctx)       // [B,S,C] heads re-merged
{
    __shared__ float lds[16 * 33];

    const int qt   = blockIdx.x % QTILES;
    const int bh   = blockIdx.x / QTILES;      // 0..15
    const int b    = bh >> 3;
    const int head = bh & 7;
    const int lane = threadIdx.x;
    const int ln   = lane & 15;
    const int kh   = lane >> 4;

    const size_t qkBase = (size_t)bh * SEQ * HDIM;
    const _Float16* Kb = K + qkBase;
    const _Float16* Vb = Vt + (size_t)bh * HDIM * SEQ;

    // Q A-fragment, resident for the whole pass
    const v16h Af = load_a_frag(Q + qkBase + (size_t)(qt * 16 + ln) * HDIM, kh);

    v8f acc0 = {}, acc1 = {};                  // ctx accum: d cols 0..15 / 16..31
    float run_m[8], run_s[8];
#pragma unroll
    for (int r = 0; r < 8; ++r) { run_m[r] = -1e30f; run_s[r] = 0.0f; }

    for (int t0 = 0; t0 < SEQ; t0 += 32) {
        // prefetch next chunk (global_prefetch_b8); speculative, OOB is dropped
        __builtin_prefetch(Kb + (size_t)(t0 + 32) * HDIM + lane * 32, 0, 1);
        __builtin_prefetch(Vb + (size_t)lane * SEQ + t0 + 32, 0, 1);

        // scores: one WMMA per 16-key tile (K-dim = d = 32)
        v16h kb0 = load_b_frag(Kb + (size_t)(t0 + ln) * HDIM, kh);
        v16h kb1 = load_b_frag(Kb + (size_t)(t0 + 16 + ln) * HDIM, kh);
        v8f z = {};
        v8f s0 = wmma_f16(Af, kb0, z);
        v8f s1 = wmma_f16(Af, kb1, z);

        // online softmax over the 32-key chunk
        float cm[8];
#pragma unroll
        for (int r = 0; r < 8; ++r) cm[r] = fmaxf(s0[r], s1[r]);
#pragma unroll
        for (int r = 0; r < 8; ++r) {
            cm[r] = fmaxf(cm[r], __shfl_xor(cm[r], 1, 32));
            cm[r] = fmaxf(cm[r], __shfl_xor(cm[r], 2, 32));
            cm[r] = fmaxf(cm[r], __shfl_xor(cm[r], 4, 32));
            cm[r] = fmaxf(cm[r], __shfl_xor(cm[r], 8, 32));
        }
#pragma unroll
        for (int r = 0; r < 8; ++r) {
            float nm   = fmaxf(run_m[r], cm[r]);
            float corr = __expf(run_m[r] - nm);
            run_m[r]   = nm;
            float p0 = __expf(s0[r] - nm);
            float p1 = __expf(s1[r] - nm);
            float ps = p0 + p1;
            ps += __shfl_xor(ps, 1, 32);
            ps += __shfl_xor(ps, 2, 32);
            ps += __shfl_xor(ps, 4, 32);
            ps += __shfl_xor(ps, 8, 32);
            run_s[r] = run_s[r] * corr + ps;
            acc0[r] *= corr;
            acc1[r] *= corr;
            lds[(r + 8 * kh) * 33 + ln]      = p0;   // D-layout -> LDS tile
            lds[(r + 8 * kh) * 33 + 16 + ln] = p1;
        }
        __syncthreads();

        // rebuild P as an A-operand fragment (lane = M) from LDS
        v16h pa;
        const float* lrow = &lds[ln * 33];
#pragma unroll
        for (int i = 0; i < 8; ++i) pa[i]     = (_Float16)lrow[8 * kh + i];
#pragma unroll
        for (int i = 0; i < 8; ++i) pa[8 + i] = (_Float16)lrow[16 + 8 * kh + i];
        __syncthreads();

        // ctx += P (16x32) x V chunk (32x32 split into two N=16 tiles)
        v16h vb0 = load_b_frag(Vb + (size_t)ln * SEQ + t0, kh);
        v16h vb1 = load_b_frag(Vb + (size_t)(16 + ln) * SEQ + t0, kh);
        acc0 = wmma_f16(pa, vb0, acc0);
        acc1 = wmma_f16(pa, vb1, acc1);
    }

    // normalize and store (heads re-merged into [B,S,C] f16)
#pragma unroll
    for (int r = 0; r < 8; ++r) {
        float inv = 1.0f / run_s[r];
        int srow  = qt * 16 + r + 8 * kh;
        _Float16* p = ctx + ((size_t)b * SEQ + srow) * C_DIM + head * HDIM;
        p[ln]      = (_Float16)(acc0[r] * inv);
        p[16 + ln] = (_Float16)(acc1[r] * inv);
    }
}

// ---------------------------------------------------------------------------
// Kernel 3: output projection  out = ctx @ Wo^T + bo   (f32 out)
// grid = MROWS/16 blocks of 8 waves; wave w owns N tile [32w, 32w+32).
// ---------------------------------------------------------------------------
__global__ __launch_bounds__(256) void out_proj_kernel(
    const _Float16* __restrict__ ctx,  // [B,S,C] f16
    const _Float16* __restrict__ wo,   // [C,C] f16
    const float* __restrict__ bo,
    float* __restrict__ out)           // [B,S,C] f32
{
    const int row0 = blockIdx.x * 16;
    const int wave = threadIdx.x >> 5;
    const int lane = threadIdx.x & 31;
    const int ln   = lane & 15;
    const int kh   = lane >> 4;
    const int n0   = wave * 32;

    const _Float16* arow  = ctx + (size_t)(row0 + ln) * C_DIM;
    const _Float16* brow0 = wo + (size_t)(n0 + ln) * C_DIM;
    const _Float16* brow1 = wo + (size_t)(n0 + 16 + ln) * C_DIM;

    v8f acc0 = {}, acc1 = {};
#pragma unroll
    for (int k0 = 0; k0 < C_DIM; k0 += 32) {
        v16h A  = load_a_frag(arow + k0, kh);
        v16h B0 = load_b_frag(brow0 + k0, kh);
        v16h B1 = load_b_frag(brow1 + k0, kh);
        acc0 = wmma_f16(A, B0, acc0);
        acc1 = wmma_f16(A, B1, acc1);
    }
    float b0 = bo[n0 + ln];
    float b1 = bo[n0 + 16 + ln];
#pragma unroll
    for (int r = 0; r < 8; ++r) {
        int row = row0 + r + 8 * kh;
        out[(size_t)row * C_DIM + n0 + ln]      = acc0[r] + b0;
        out[(size_t)row * C_DIM + n0 + 16 + ln] = acc1[r] + b1;
    }
}

// ---------------------------------------------------------------------------
extern "C" void kernel_launch(void* const* d_in, const int* in_sizes, int n_in,
                              void* d_out, int out_size, void* d_ws, size_t ws_size,
                              hipStream_t stream) {
    const float* x  = (const float*)d_in[0];   // [2,64,48,256]
    const float* Wq = (const float*)d_in[1];
    const float* Wk = (const float*)d_in[2];
    const float* Wv = (const float*)d_in[3];
    const float* qb = (const float*)d_in[4];
    const float* vb = (const float*)d_in[5];
    const float* Wo = (const float*)d_in[6];
    const float* bo = (const float*)d_in[7];
    float* out = (float*)d_out;

    const int NX = MROWS * C_DIM;        // 1,572,864
    const int NW = C_DIM * C_DIM;        // 65,536

    _Float16* ws  = (_Float16*)d_ws;
    _Float16* xh  = ws;            ws += NX;
    _Float16* wqh = ws;            ws += NW;
    _Float16* wkh = ws;            ws += NW;
    _Float16* wvh = ws;            ws += NW;
    _Float16* woh = ws;            ws += NW;
    _Float16* Qh  = ws;            ws += NX;   // [B,h,S,d]
    _Float16* Kh  = ws;            ws += NX;   // [B,h,S,d]
    _Float16* Vth = ws;            ws += NX;   // [B,h,d,S]
    _Float16* Ctx = ws;            ws += NX;   // [B,S,C]

    cvt_f16_kernel<<<(NX + 255) / 256, 256, 0, stream>>>(x, xh, NX);
    cvt_f16_kernel<<<(NW + 255) / 256, 256, 0, stream>>>(Wq, wqh, NW);
    cvt_f16_kernel<<<(NW + 255) / 256, 256, 0, stream>>>(Wk, wkh, NW);
    cvt_f16_kernel<<<(NW + 255) / 256, 256, 0, stream>>>(Wv, wvh, NW);
    cvt_f16_kernel<<<(NW + 255) / 256, 256, 0, stream>>>(Wo, woh, NW);

    qkv_rope_kernel<<<dim3(MROWS / 16, 1, 3), 256, 0, stream>>>(
        xh, wqh, wkh, wvh, qb, vb, Qh, Kh, Vth);

    flash_attn_kernel<<<BATCH * NHEADS * QTILES, 32, 0, stream>>>(Qh, Kh, Vth, Ctx);

    out_proj_kernel<<<MROWS / 16, 256, 0, stream>>>(Ctx, woh, bo, out);
}
